// Recurrence_70385924047001
// MI455X (gfx1250) — compile-verified, matrix operand order checked by reading, not run on hardware
//
#include <hip/hip_runtime.h>

typedef __attribute__((ext_vector_type(16))) _Float16 v16h;
typedef __attribute__((ext_vector_type(8)))  float    v8f;

#define T_   32
#define N_   64
#define NL   64
#define E_   128
#define NTOK 4
#define CH   64
#define D_   16
#define HW   16
#define INVN 32
#define RH   64
#define LE   64
#define HD   512
#define A_   16
#define NE   4
#define ZIN  320      // E + CH + RH + LE
#define KW   9216     // CH*D*3*3
#define KP   160      // padded per-channel K for conv GEMM (144 -> 160)
#define G3E  384      // 3*E
#define OUTW 663      // 1+16+1+128+512+1+1+2+1

__device__ __forceinline__ float sigmoidf(float x) { return 1.f / (1.f + __expf(-x)); }

__device__ __forceinline__ v8f wmma16(v16h a, v16h b, v8f c) {
  return __builtin_amdgcn_wmma_f32_16x16x32_f16(false, a, false, b, (short)0, c, false, false);
}

// Fragment of a row-major f16 matrix: lane holds row (row0 + lane%16), K-pattern per ISA layout.
// Used for A tiles, and for B tiles when the B matrix is W^T with W stored row-major [N][K].
__device__ __forceinline__ v16h frag_row(const _Float16* base, int ld, int row0, int k0) {
  int lane = threadIdx.x & 31;
  const _Float16* p = base + (size_t)(row0 + (lane & 15)) * ld + k0 + ((lane >> 4) << 3);
  v16h f;
#pragma unroll
  for (int i = 0; i < 8; ++i) { f[i] = p[i]; f[i + 8] = p[i + 16]; }
  return f;
}

// ---------------- setup kernels ----------------

__global__ void k_f2h(const float* __restrict__ s, _Float16* __restrict__ d, int n) {
  int i = blockIdx.x * blockDim.x + threadIdx.x;
  if (i < n) d[i] = (_Float16)s[i];
}

// transpose + convert: s is [R][C] row-major f32 -> d is [C][R] row-major f16
__global__ void k_f2h_T(const float* __restrict__ s, _Float16* __restrict__ d, int R, int C) {
  int i = blockIdx.x * blockDim.x + threadIdx.x;
  if (i < R * C) {
    int r = i / C, c = i % C;
    d[(size_t)c * R + r] = (_Float16)s[i];
  }
}

__global__ void k_h_zero(_Float16* __restrict__ d, int n) {
  int i = blockIdx.x * blockDim.x + threadIdx.x;
  if (i < n) d[i] = (_Float16)0.f;
}

__global__ void k_init_p(const int* __restrict__ p0, int* __restrict__ pb) {
  int i = threadIdx.x;
  if (i < N_) pb[i] = p0[i];
}

// M[n,l,:] = sum_tok emb_task[lines[n,l,tok],:]
__global__ void k_embed(const int* __restrict__ lines, const float* __restrict__ emb,
                        float* __restrict__ Mf, _Float16* __restrict__ Mh) {
  int row = blockIdx.x;          // n*NL + l
  int e = threadIdx.x;           // 0..127
  const int* ln = lines + row * NTOK;
  float s = 0.f;
#pragma unroll
  for (int t = 0; t < NTOK; ++t) s += emb[ln[t] * E_ + e];
  Mf[row * E_ + e] = s;
  Mh[row * E_ + e] = (_Float16)s;
}

// Gi[dir][n,s,:] = M[n,s] @ wi.T + bi  -- [4096,128] x [128,384] WMMA GEMM, both dirs
__global__ void __launch_bounds__(256) k_gi(const _Float16* __restrict__ Mh,
                                            const _Float16* __restrict__ wif,
                                            const _Float16* __restrict__ wib,
                                            const float* __restrict__ bif,
                                            const float* __restrict__ bib,
                                            float* __restrict__ Gf, float* __restrict__ Gb) {
  int tid = blockIdx.x * 8 + (threadIdx.x >> 5);  // tile id: 2*256*24 = 12288
  int dir = tid / (256 * 24);
  int r = tid % (256 * 24);
  int mt = r / 24, nt = r % 24;
  const _Float16* w = dir ? wib : wif;
  const float* bi = dir ? bib : bif;
  float* out = dir ? Gb : Gf;
  int lane = threadIdx.x & 31;
  int col = nt * 16 + (lane & 15);
  float bias = bi[col];
  v8f acc;
#pragma unroll
  for (int i = 0; i < 8; ++i) acc[i] = bias;
#pragma unroll
  for (int ks = 0; ks < 4; ++ks) {
    v16h a = frag_row(Mh, E_, mt * 16, ks * 32);
    v16h b = frag_row(w, E_, nt * 16, ks * 32);   // B = wi^T, wi row-major [384][128]
    acc = wmma16(a, b, acc);
  }
  int rbase = mt * 16 + ((lane >> 4) << 3);
#pragma unroll
  for (int i = 0; i < 8; ++i) out[(size_t)(rbase + i) * G3E + col] = acc[i];
}

// inv=relu(inv@inv_w+b) and pa=embedding-bag for all (t,n)
__global__ void k_pre(const float* __restrict__ inv, const int* __restrict__ pa,
                      const float* __restrict__ invw, const float* __restrict__ invb,
                      const float* __restrict__ embl,
                      float* __restrict__ invpre, float* __restrict__ papre) {
  int tn = blockIdx.x;
  int j = threadIdx.x;           // 0..63
  const float* it = inv + tn * INVN;
  float s = invb[j];
  for (int k = 0; k < INVN; ++k) s += it[k] * invw[k * RH + j];
  invpre[tn * RH + j] = fmaxf(s, 0.f);
  const int* pt = pa + tn * 4;
  float s2 = 0.f;
#pragma unroll
  for (int k = 0; k < 4; ++k) s2 += embl[pt[k] * LE + j];
  papre[tn * LE + j] = s2;
}

// ---------------- per-timestep kernels ----------------

// Bidirectional GRU recurrence. 8 blocks: (dir, 16-example group). Wh B-fragments live
// in VGPRs for all 64 steps; h in LDS f16; per-step [16,128]x[128,384] WMMA GEMM.
// Gathered Gi rows are prefetched (global_prefetch_b8) before the WMMA phase.
__global__ void __launch_bounds__(256) k_gru(const float* __restrict__ Gif,
                                             const float* __restrict__ Gib,
                                             const _Float16* __restrict__ whf,
                                             const _Float16* __restrict__ whb,
                                             const float* __restrict__ bhf,
                                             const float* __restrict__ bhb,
                                             const int* __restrict__ pin,
                                             float* __restrict__ G) {
  __shared__ _Float16 h16[16][E_];
  __shared__ float gh[16][G3E];
  __shared__ int ps[16];
  int dir = blockIdx.x & 1, grp = blockIdx.x >> 1;
  const float* Gi = dir ? Gib : Gif;
  const _Float16* wh = dir ? whb : whf;
  const float* bh = dir ? bhb : bhf;
  int tid = threadIdx.x, lane = tid & 31, wv = tid >> 5;
  if (tid < 16) ps[tid] = pin[grp * 16 + tid];
  for (int i = tid; i < 16 * E_; i += 256) ((_Float16*)h16)[i] = (_Float16)0.f;
  v16h bf[3][4];                 // resident Wh fragments: 3 n-tiles x 4 k-steps per wave
#pragma unroll
  for (int j = 0; j < 3; ++j)
#pragma unroll
    for (int ks = 0; ks < 4; ++ks)
      bf[j][ks] = frag_row(wh, E_, (wv * 3 + j) * 16, ks * 32);
  __syncthreads();
  for (int l = 0; l < NL; ++l) {
    int pos = dir ? (NL - 1 - l) : l;
    // prefetch the 16 gathered Gi rows (12 cachelines each) behind the WMMA phase
    {
      int ex = tid >> 4, lo = tid & 15;
      if (lo < 12) {
        int src = (ps[ex] + pos) & (NL - 1);
        __builtin_prefetch(Gi + (size_t)((grp * 16 + ex) * NL + src) * G3E + lo * 32, 0, 0);
      }
    }
    // gh = h @ Wh^T + bh
#pragma unroll
    for (int j = 0; j < 3; ++j) {
      int col = (wv * 3 + j) * 16 + (lane & 15);
      float bias = bh[col];
      v8f acc;
#pragma unroll
      for (int i = 0; i < 8; ++i) acc[i] = bias;
#pragma unroll
      for (int ks = 0; ks < 4; ++ks) {
        v16h a = frag_row(&h16[0][0], E_, 0, ks * 32);
        acc = wmma16(a, bf[j][ks], acc);
      }
      int rbase = ((lane >> 4) << 3);
#pragma unroll
      for (int i = 0; i < 8; ++i) gh[rbase + i][col] = acc[i];
    }
    __syncthreads();
#pragma unroll
    for (int it = 0; it < 8; ++it) {
      int idx = it * 256 + tid;            // 16 examples x 128 features
      int ex = idx >> 7, e = idx & 127;
      int n = grp * 16 + ex;
      int src = (ps[ex] + pos) & (NL - 1);
      const float* gi = Gi + (size_t)(n * NL + src) * G3E;
      float rr = sigmoidf(gi[e] + gh[ex][e]);
      float zz = sigmoidf(gi[E_ + e] + gh[ex][E_ + e]);
      float nn = tanhf(gi[2 * E_ + e] + rr * gh[ex][2 * E_ + e]);
      float hp = (float)h16[ex][e];
      float hv = (1.f - zz) * nn + zz * hp;
      h16[ex][e] = (_Float16)hv;
      G[((size_t)(n * NL + pos) * 2 + dir) * E_ + e] = hv;
    }
    __syncthreads();
  }
}

// beta head + cumprod attention -> P[n, 2nl, NE]
__global__ void k_beta(const float* __restrict__ G, const float* __restrict__ bw,
                       const float* __restrict__ bb, float* __restrict__ P) {
  __shared__ float zl[128][NE];
  int n = blockIdx.x, tid = threadIdx.x;   // 256 threads
#pragma unroll
  for (int h = 0; h < 2; ++h) {
    int oidx = h * 256 + tid;              // 0..511
    int q = oidx >> 2, j = oidx & 3;
    int d = q & 1, lq = q >> 1;
    int lsrc = d ? (NL - 1 - lq) : lq;
    const float* g = G + ((size_t)(n * NL + lsrc) * 2 + d) * E_;
    float s = bb[j];
    for (int e = 0; e < E_; ++e) s += g[e] * bw[e * NE + j];
    float v = sigmoidf(s);
    if (q == 0) v = 0.f;                   // (1-last)[::-1] mask
    if (q == 127) v = 0.f;                 // zero_last
    zl[q][j] = v;
  }
  __syncthreads();
  if (tid < NE) {
    int j = tid;
    float c = 1.f;                         // cumprod of (1 - rolled zero_last)
    for (int q = 0; q < 128; ++q) {
      float bfin = (q == 127) ? 1.f : zl[q][j];
      int o = (q & 1) ? (63 - (q >> 1)) : (64 + (q >> 1));
      P[(size_t)(n * 128 + o) * NE + j] = bfin * c;
      c *= (1.f - zl[q][j]);
    }
  }
}

// Kern[n,c,0:144] = (M[n,p[n]] @ kernel_w + kernel_b) -> f16, rows padded to 160
// ([64,128]x[128,9216] WMMA, B from pre-transposed kwT[9216][128])
__global__ void __launch_bounds__(256) k_kern(const _Float16* __restrict__ Mh,
                                              const _Float16* __restrict__ kwT,
                                              const float* __restrict__ kb,
                                              const int* __restrict__ pin,
                                              _Float16* __restrict__ Kern) {
  int tid = blockIdx.x * 8 + (threadIdx.x >> 5);  // 4*576 = 2304 tiles
  int mt = tid / 576, nt = tid % 576;
  int lane = threadIdx.x & 31;
  int col = nt * 16 + (lane & 15);                // flattened c*144 + dkk
  float bias = kb[col];
  v8f acc;
#pragma unroll
  for (int i = 0; i < 8; ++i) acc[i] = bias;
  int nrow = mt * 16 + (lane & 15);
  const _Float16* arow = Mh + (size_t)(nrow * NL + pin[nrow]) * E_;  // pointer-gathered A
#pragma unroll
  for (int ks = 0; ks < 4; ++ks) {
    int kb0 = ks * 32 + ((lane >> 4) << 3);
    v16h a;
#pragma unroll
    for (int i = 0; i < 8; ++i) { a[i] = arow[kb0 + i]; a[i + 8] = arow[kb0 + 16 + i]; }
    v16h b = frag_row(kwT, E_, nt * 16, ks * 32);  // contiguous: kwT row = kernel_w column
    acc = wmma16(a, b, acc);
  }
  int rbase = mt * 16 + ((lane >> 4) << 3);
  int c = col / 144, k2 = col % 144;
#pragma unroll
  for (int i = 0; i < 8; ++i)
    Kern[(size_t)(rbase + i) * (CH * KP) + c * KP + k2] = (_Float16)acc[i];
}

// Per-example dynamic conv as im2col WMMA GEMM [256 spatial,160] x [160,64ch].
// obs tile staged into LDS via async global->LDS copy (ASYNCcnt), im2col via LDS LUT,
// bias folded into C init, ReLU + spatial-sum via ds_add_f32.
__global__ void __launch_bounds__(256) k_conv(const float* __restrict__ obs,
                                              const _Float16* __restrict__ Kern,
                                              const float* __restrict__ cbias,
                                              float* __restrict__ h1, int t) {
  __shared__ float stg[D_ * HW * HW];           // async-staged raw obs (f32)
  __shared__ _Float16 op[D_][HW + 2][HW + 2];   // zero-padded f16 input
  __shared__ int lut[KP];                       // im2col offset table
  __shared__ float h1s[CH];
  int n = blockIdx.x, tid = threadIdx.x, lane = tid & 31, wv = tid >> 5;

  // kick off async LDS fill of the obs tile (1024 x b128 = 16KB)
  const float* ob = obs + (size_t)(t * N_ + n) * (D_ * HW * HW);
  unsigned lbase = (unsigned)(uintptr_t)&stg[0];
  for (int i = tid; i < (D_ * HW * HW) / 4; i += 256) {
    unsigned ldst = lbase + i * 16;
    const float* gsrc = ob + i * 4;
    asm volatile("global_load_async_to_lds_b128 %0, %1, off"
                 :: "v"(ldst), "v"(gsrc) : "memory");
  }
  // overlap: zero padded buffer, build LUT, zero h1 accumulators
  for (int i = tid; i < D_ * (HW + 2) * (HW + 2); i += 256) ((_Float16*)op)[i] = (_Float16)0.f;
  if (tid < KP) {
    int kk = tid, off = -1;
    if (kk < 144) {
      int d = kk / 9, r9 = kk % 9;
      off = d * ((HW + 2) * (HW + 2)) + (r9 / 3) * (HW + 2) + (r9 % 3);
    }
    lut[tid] = off;
  }
  if (tid < CH) h1s[tid] = 0.f;
  asm volatile("s_wait_asynccnt 0x0" ::: "memory");
  __syncthreads();
  // f32 -> f16 into padded layout
  for (int i = tid; i < D_ * HW * HW; i += 256) {
    int dd = i >> 8, rr = (i >> 4) & 15, cc = i & 15;
    op[dd][rr + 1][cc + 1] = (_Float16)stg[i];
  }
  __syncthreads();

  const _Float16* opf = &op[0][0][0];
  const _Float16* kn = Kern + (size_t)n * (CH * KP);
  for (int mi = 0; mi < 2; ++mi) {
    int mt = wv * 2 + mi;
    int m = mt * 16 + (lane & 15);
    int base = (m >> 4) * (HW + 2) + (m & 15);   // x*18 + y
    for (int ct = 0; ct < 4; ++ct) {
      int col = ct * 16 + (lane & 15);
      float bias = cbias[col];
      v8f acc;
#pragma unroll
      for (int i = 0; i < 8; ++i) acc[i] = bias;
      const _Float16* bch = kn + col * KP;
      for (int ks = 0; ks < 5; ++ks) {
        int kb0 = ks * 32 + ((lane >> 4) << 3);
        const _Float16* bp = bch + kb0;           // contiguous, pad region pre-zeroed
        v16h a, b;
#pragma unroll
        for (int i = 0; i < 8; ++i) { b[i] = bp[i]; b[i + 8] = bp[i + 16]; }
#pragma unroll
        for (int i = 0; i < 16; ++i) {
          int kk = kb0 + (i & 7) + ((i >> 3) << 4);
          int off = lut[kk];
          a[i] = (off >= 0) ? opf[base + off] : (_Float16)0.f;
        }
        acc = wmma16(a, b, acc);
      }
      float part = 0.f;
#pragma unroll
      for (int i = 0; i < 8; ++i) part += fmaxf(acc[i], 0.f);
      atomicAdd(&h1s[col], part);
    }
  }
  __syncthreads();
  if (tid < CH) h1[n * CH + tid] = h1s[tid];
}

// Assemble z_in = [m | h1 | inv | pa]
__global__ void k_zin(const float* __restrict__ Mf, const float* __restrict__ h1,
                      const float* __restrict__ invpre, const float* __restrict__ papre,
                      const int* __restrict__ pin, _Float16* __restrict__ zin16,
                      float* __restrict__ zinf, int t) {
  int n = blockIdx.x, j = threadIdx.x;   // 320 threads
  float v;
  if (j < 128)       v = Mf[(size_t)(n * NL + pin[n]) * E_ + j];
  else if (j < 192)  v = h1[n * CH + (j - 128)];
  else if (j < 256)  v = invpre[(size_t)(t * N_ + n) * RH + (j - 192)];
  else               v = papre[(size_t)(t * N_ + n) * LE + (j - 256)];
  zinf[n * ZIN + j] = v;
  zin16[n * ZIN + j] = (_Float16)v;
}

// z1 = relu(z_in @ zeta1_w + b)  ([64,320]x[320,512] WMMA, B from zwT[512][320]),
// written directly into d_out
__global__ void __launch_bounds__(256) k_zeta(const _Float16* __restrict__ zin16,
                                              const _Float16* __restrict__ zwT,
                                              const float* __restrict__ zb,
                                              float* __restrict__ out, int t) {
  int tid = blockIdx.x * 8 + (threadIdx.x >> 5);  // 4*32 = 128 tiles
  int mt = tid >> 5, nt = tid & 31;
  int lane = threadIdx.x & 31;
  int col = nt * 16 + (lane & 15);
  float bias = zb[col];
  v8f acc;
#pragma unroll
  for (int i = 0; i < 8; ++i) acc[i] = bias;
#pragma unroll
  for (int ks = 0; ks < 10; ++ks) {
    v16h a = frag_row(zin16, ZIN, mt * 16, ks * 32);
    v16h b = frag_row(zwT, ZIN, nt * 16, ks * 32);  // contiguous: zwT row = zeta1_w column
    acc = wmma16(a, b, acc);
  }
  int rbase = mt * 16 + ((lane >> 4) << 3);
#pragma unroll
  for (int i = 0; i < 8; ++i) {
    int n = rbase + i;
    out[(size_t)(t * N_ + n) * OUTW + 146 + col] = fmaxf(acc[i], 0.f);
  }
}

// heads: actor softmax*mask, upsilon softmax, d_probs, dgate, critic, argmaxes, pointer update
__global__ void k_out(const float* __restrict__ zinf, const float* __restrict__ P,
                      const float* __restrict__ amask,
                      const float* __restrict__ aw, const float* __restrict__ ab,
                      const float* __restrict__ uw, const float* __restrict__ ub,
                      const float* __restrict__ dw, const float* __restrict__ db,
                      const float* __restrict__ cw, const float* __restrict__ cb,
                      const int* __restrict__ pin, int* __restrict__ pout,
                      float* __restrict__ out, int t) {
  __shared__ float sl[A_], sp[A_], su[NE], sd[128], sdg[2], red[64], svu[8];
  int n = blockIdx.x, tid = threadIdx.x;   // 64 threads
  float* orow = out + (size_t)(t * N_ + n) * OUTW;
  const float* z1 = orow + 146;
  const float* zi = zinf + n * ZIN;
  if (tid < A_) {
    float s = ab[tid];
    for (int e = 0; e < HD; ++e) s += z1[e] * aw[e * A_ + tid];
    sl[tid] = s;
  }
  if (tid >= 16 && tid < 20) {
    int j = tid - 16;
    float s = ub[j];
    for (int e = 0; e < ZIN; ++e) s += zi[e] * uw[e * NE + j];
    svu[j] = s;
  }
  if (tid >= 20 && tid < 22) {
    int j = tid - 20;
    float s = db[j];
    for (int e = 0; e < ZIN; ++e) s += zi[e] * dw[e * 2 + j];
    svu[4 + j] = s;
  }
  {
    float s = 0.f;
    for (int e = tid; e < HD; e += 64) s += z1[e] * cw[e];
    red[tid] = s;
  }
  __syncthreads();
  if (tid == 0) {
    float mx = sl[0];
    for (int j = 1; j < A_; ++j) mx = fmaxf(mx, sl[j]);
    float sum = 0.f;
    for (int j = 0; j < A_; ++j) { float e = __expf(sl[j] - mx); sp[j] = e; sum += e; }
    const float* msk = amask + (size_t)(t * N_ + n) * A_;
    float invs = 1.f / sum;
    for (int j = 0; j < A_; ++j) sp[j] = sp[j] * invs * msk[j];
  }
  if (tid == 1) {
    float mx = fmaxf(fmaxf(svu[0], svu[1]), fmaxf(svu[2], svu[3]));
    float e0 = __expf(svu[0] - mx), e1 = __expf(svu[1] - mx);
    float e2 = __expf(svu[2] - mx), e3 = __expf(svu[3] - mx);
    float s = e0 + e1 + e2 + e3;
    su[0] = e0 / s; su[1] = e1 / s; su[2] = e2 / s; su[3] = e3 / s;
  }
  if (tid == 2) {
    float mx = fmaxf(svu[4], svu[5]);
    float e0 = __expf(svu[4] - mx), e1 = __expf(svu[5] - mx);
    sdg[0] = e0 / (e0 + e1); sdg[1] = e1 / (e0 + e1);
  }
  __syncthreads();
#pragma unroll
  for (int h = 0; h < 2; ++h) {
    int l = h * 64 + tid;
    const float* pr = P + (size_t)(n * 128 + l) * NE;
    float s = 0.f;
#pragma unroll
    for (int j = 0; j < NE; ++j) s += pr[j] * su[j];
    sd[l] = s;
  }
  __syncthreads();
  if (tid == 0) {
    float v = 0.f;
    for (int i = 0; i < 64; ++i) v += red[i];
    v += cb[0];
    int a = 0; float am = sp[0];
    for (int j = 1; j < A_; ++j) if (sp[j] > am) { am = sp[j]; a = j; }
    int dg = (sdg[1] > sdg[0]) ? 1 : 0;
    int dsel;
    if (dg) {
      int d0 = 0; float dm = sd[0];
      for (int l = 1; l < 128; ++l) if (sd[l] > dm) { dm = sd[l]; d0 = l; }
      dsel = d0;
    } else dsel = NL;                        // argmax(one_hot at 64)
    int pn = pin[n] + dsel - NL;
    pn = pn < 0 ? 0 : (pn > NL - 1 ? NL - 1 : pn);
    pout[n] = pn;
    orow[0]   = (float)a;
    orow[17]  = (float)dsel;
    orow[658] = (float)pn;
    orow[659] = v;
    orow[660] = sdg[0];
    orow[661] = sdg[1];
    orow[662] = (float)dg;
  }
  __syncthreads();
  if (tid < A_) orow[1 + tid] = sp[tid];
#pragma unroll
  for (int h = 0; h < 2; ++h) { int l = h * 64 + tid; orow[18 + l] = sd[l]; }
}

// ---------------- host launcher ----------------

extern "C" void kernel_launch(void* const* d_in, const int* in_sizes, int n_in,
                              void* d_out, int out_size, void* d_ws, size_t ws_size,
                              hipStream_t stream) {
  (void)in_sizes; (void)n_in; (void)out_size; (void)ws_size;
  const int*   lines          = (const int*)d_in[0];
  const float* obs            = (const float*)d_in[1];
  const float* inventory      = (const float*)d_in[2];
  const int*   partial_action = (const int*)d_in[3];
  const float* action_mask    = (const float*)d_in[4];
  const int*   p0             = (const int*)d_in[5];
  const float* emb_task       = (const float*)d_in[6];
  const float* emb_lower      = (const float*)d_in[7];
  const float* wi_f = (const float*)d_in[8];
  const float* wh_f = (const float*)d_in[9];
  const float* bi_f = (const float*)d_in[10];
  const float* bh_f = (const float*)d_in[11];
  const float* wi_b = (const float*)d_in[12];
  const float* wh_b = (const float*)d_in[13];
  const float* bi_b = (const float*)d_in[14];
  const float* bh_b = (const float*)d_in[15];
  const float* beta_w  = (const float*)d_in[16];
  const float* beta_b  = (const float*)d_in[17];
  const float* kernel_w = (const float*)d_in[18];
  const float* kernel_b = (const float*)d_in[19];
  const float* conv_bias = (const float*)d_in[20];
  const float* inv_w = (const float*)d_in[21];
  const float* inv_b = (const float*)d_in[22];
  const float* zeta1_w = (const float*)d_in[23];
  const float* zeta1_b = (const float*)d_in[24];
  const float* actor_w = (const float*)d_in[25];
  const float* actor_b = (const float*)d_in[26];
  const float* upsilon_w = (const float*)d_in[27];
  const float* upsilon_b = (const float*)d_in[28];
  const float* dgate_w = (const float*)d_in[29];
  const float* dgate_b = (const float*)d_in[30];
  const float* critic_w = (const float*)d_in[31];
  const float* critic_b = (const float*)d_in[32];
  float* out = (float*)d_out;

  char* wp = (char*)d_ws;
  auto carve = [&](size_t bytes) -> void* {
    void* p = (void*)wp;
    wp += (bytes + 255) & ~(size_t)255;
    return p;
  };
  float*    Mf    = (float*)carve((size_t)N_ * NL * E_ * 4);
  _Float16* Mh    = (_Float16*)carve((size_t)N_ * NL * E_ * 2);
  _Float16* wi16f = (_Float16*)carve((size_t)G3E * E_ * 2);
  _Float16* wi16b = (_Float16*)carve((size_t)G3E * E_ * 2);
  _Float16* wh16f = (_Float16*)carve((size_t)G3E * E_ * 2);
  _Float16* wh16b = (_Float16*)carve((size_t)G3E * E_ * 2);
  _Float16* kwT   = (_Float16*)carve((size_t)E_ * KW * 2);     // kernel_w^T [9216][128]
  _Float16* zwT   = (_Float16*)carve((size_t)ZIN * HD * 2);    // zeta1_w^T  [512][320]
  float*    Gif   = (float*)carve((size_t)N_ * NL * G3E * 4);
  float*    Gib   = (float*)carve((size_t)N_ * NL * G3E * 4);
  float*    G     = (float*)carve((size_t)N_ * NL * 2 * E_ * 4);
  float*    P     = (float*)carve((size_t)N_ * 128 * NE * 4);
  _Float16* Kern  = (_Float16*)carve((size_t)N_ * CH * KP * 2);
  float*    h1    = (float*)carve((size_t)N_ * CH * 4);
  float*    invpre = (float*)carve((size_t)T_ * N_ * RH * 4);
  float*    papre  = (float*)carve((size_t)T_ * N_ * LE * 4);
  _Float16* zin16 = (_Float16*)carve((size_t)N_ * ZIN * 2);
  float*    zinf  = (float*)carve((size_t)N_ * ZIN * 4);
  int*      pbuf  = (int*)carve((size_t)2 * N_ * 4);

  k_init_p<<<1, 64, 0, stream>>>(p0, pbuf);
  k_f2h<<<(G3E * E_ + 255) / 256, 256, 0, stream>>>(wi_f, wi16f, G3E * E_);
  k_f2h<<<(G3E * E_ + 255) / 256, 256, 0, stream>>>(wi_b, wi16b, G3E * E_);
  k_f2h<<<(G3E * E_ + 255) / 256, 256, 0, stream>>>(wh_f, wh16f, G3E * E_);
  k_f2h<<<(G3E * E_ + 255) / 256, 256, 0, stream>>>(wh_b, wh16b, G3E * E_);
  k_f2h_T<<<(E_ * KW + 255) / 256, 256, 0, stream>>>(kernel_w, kwT, E_, KW);
  k_f2h_T<<<(ZIN * HD + 255) / 256, 256, 0, stream>>>(zeta1_w, zwT, ZIN, HD);
  k_h_zero<<<(N_ * CH * KP + 255) / 256, 256, 0, stream>>>(Kern, N_ * CH * KP);
  k_embed<<<N_ * NL, E_, 0, stream>>>(lines, emb_task, Mf, Mh);
  k_gi<<<1536, 256, 0, stream>>>(Mh, wi16f, wi16b, bi_f, bi_b, Gif, Gib);
  k_pre<<<T_ * N_, 64, 0, stream>>>(inventory, partial_action, inv_w, inv_b, emb_lower,
                                    invpre, papre);

  for (int t = 0; t < T_; ++t) {
    int* pin  = pbuf + (t & 1) * N_;
    int* pout = pbuf + ((t + 1) & 1) * N_;
    k_gru<<<8, 256, 0, stream>>>(Gif, Gib, wh16f, wh16b, bh_f, bh_b, pin, G);
    k_kern<<<288, 256, 0, stream>>>(Mh, kwT, kernel_b, pin, Kern);
    k_beta<<<64, 256, 0, stream>>>(G, beta_w, beta_b, P);
    k_conv<<<64, 256, 0, stream>>>(obs, Kern, conv_bias, h1, t);
    k_zin<<<64, ZIN, 0, stream>>>(Mf, h1, invpre, papre, pin, zin16, zinf, t);
    k_zeta<<<16, 256, 0, stream>>>(zin16, zwT, zeta1_b, out, t);
    k_out<<<64, 64, 0, stream>>>(zinf, P, action_mask, actor_w, actor_b,
                                 upsilon_w, upsilon_b, dgate_w, dgate_b,
                                 critic_w, critic_b, pin, pout, out, t);
  }
}
